// GraphAttentionLayer_88648124989941
// MI455X (gfx1250) — matrix operand order repaired and checked by reading, not compiled, not run on hardware
//
#include <hip/hip_runtime.h>
#include <hip/hip_bf16.h>
#include <float.h>
#include <math.h>
#include <stdint.h>

// ---- problem constants ----
#define Nn    4096
#define FINd  512
#define FOUTd 64
#define CCd   3
#define NP    4099              // N + CC
#define NTd   4224              // padded: divisible by 128 (macro tile) and 32 (K step)
#define KNNd  5
#define TOPKd 6
#define ALPHAd 0.2f
#define EPSd  9e-15f
#define NEGd  (-9.0e15f)

typedef __attribute__((ext_vector_type(16))) __bf16 v16bf;
typedef __attribute__((ext_vector_type(8)))  float  v8f;
typedef __attribute__((ext_vector_type(4)))  unsigned int v4u;

union FragU { v4u u[2]; v16bf v; };
union Pack8 { uint4 u4; __bf16 h[8]; };

// Tensor Data Mover availability (device pass only; host pass sees fallback)
#if defined(__AMDGCN__) && __has_builtin(__builtin_amdgcn_tensor_load_to_lds) && \
    __has_builtin(__builtin_amdgcn_s_wait_tensorcnt)
#define USE_TDM 1
typedef unsigned int u32x4 __attribute__((ext_vector_type(4)));
typedef int          i32x4 __attribute__((ext_vector_type(4)));
typedef int          i32x8 __attribute__((ext_vector_type(8)));

// Issue one TDM 2D tile load: 32 bf16 (dim0) x 128 rows (dim1), row stride = stride_elems.
// Descriptor bit layout per CDNA5 ISA.
__device__ __forceinline__ void tdm_issue(const __bf16* gsrc, unsigned lds_off,
                                          int stride_elems) {
  unsigned long long ga = (unsigned long long)(uintptr_t)gsrc;
  u32x4 g0 = { 1u, lds_off, (unsigned)(ga & 0xFFFFFFFFull),
               (unsigned)((ga >> 32) & 0x01FFFFFFull) | (2u << 30) };
  i32x8 g1 = { (int)0x00010000u,                                  // data_size = 2B
               (int)(((unsigned)stride_elems & 0xFFFFu) << 16),   // tensor_dim0 lo16
               (int)(((unsigned)stride_elems & 0xFFFFu) << 16),   // tensor_dim1 lo16
               (int)(32u << 16),                                  // tile_dim0 = 32
               (int)128,                                          // tile_dim1 = 128
               (int)stride_elems,                                 // tensor_dim0_stride
               0, 0 };
  i32x4 z4 = { 0, 0, 0, 0 };
#if __clang_major__ >= 23
  i32x8 z8 = { 0, 0, 0, 0, 0, 0, 0, 0 };
  __builtin_amdgcn_tensor_load_to_lds(g0, g1, z4, z4, z8, 0);
#else
  __builtin_amdgcn_tensor_load_to_lds(g0, g1, z4, z4, 0);
#endif
}
#else
#define USE_TDM 0
#endif

// 8 WMMAs for one 32-wide K step; As/Bs are [128][32] bf16 tiles in LDS ([m][k], [n][k]).
__device__ __forceinline__ void mma_step(const __bf16* Asb, const __bf16* Bsb,
                                         int wm, int wn, int lane, int hi8, int koffB,
                                         v8f (&acc)[2][4]) {
  v16bf afr[2], bfr[4];
#pragma unroll
  for (int ms = 0; ms < 2; ++ms) {
    int rl = wm * 32 + ms * 16 + (lane & 15);
    FragU f;
    f.u[0] = *(const v4u*)&Asb[rl * 32 + hi8];
    f.u[1] = *(const v4u*)&Asb[rl * 32 + 16 + hi8];
    afr[ms] = f.v;
  }
#pragma unroll
  for (int ns = 0; ns < 4; ++ns) {
    int cl = wn * 64 + ns * 16 + (lane & 15);
    FragU f;
    f.u[0] = *(const v4u*)&Bsb[cl * 32 + hi8];
    f.u[1] = *(const v4u*)&Bsb[cl * 32 + 16 + hi8];
    bfr[ns] = f.v;
  }
#pragma unroll
  for (int ms = 0; ms < 2; ++ms)
#pragma unroll
    for (int ns = 0; ns < 4; ++ns)
      acc[ms][ns] = __builtin_amdgcn_wmma_f32_16x16x32_bf16(
          false, afr[ms], false, bfr[ns], (short)0, acc[ms][ns], false, false);
}

// ---------------- simple elementwise / reduction kernels ----------------

__global__ void k_sq(const float* __restrict__ h, float* __restrict__ sq) {
  int i = blockIdx.x * blockDim.x + threadIdx.x;
  if (i >= Nn) return;
  const float* r = h + (size_t)i * FINd;
  float s = 0.f;
  for (int f = 0; f < FINd; ++f) { float v = r[f]; s += v * v; }
  sq[i] = s;
}

// h -> bf16 copy (enables TDM staging of both Gram operands)
__global__ void k_hb(const float* __restrict__ h, __bf16* __restrict__ hb) {
  size_t idx = (size_t)blockIdx.x * blockDim.x + threadIdx.x;   // Nn*FINd/8 threads
  size_t base = idx * 8;
  Pack8 pk;
#pragma unroll
  for (int e = 0; e < 8; ++e) pk.h[e] = (__bf16)h[base + e];
  *(uint4*)&hb[base] = pk.u4;
}

// Wh = h @ W  via bf16 WMMA (one wave per 16x16 C tile; small GEMM)
__global__ void k_whgemm(const float* __restrict__ h, const float* __restrict__ W,
                         float* __restrict__ Wh) {
  int t = blockIdx.x * 4 + (threadIdx.x >> 5);
  if (t >= 256 * 4) return;
  int lane = threadIdx.x & 31;
  int mt = t >> 2, ntile = t & 3;
  const float* arow = h + (size_t)(mt * 16 + (lane & 15)) * FINd;
  int j = ntile * 16 + (lane & 15);
  int hi8 = (lane >= 16) ? 8 : 0;
  int koff = (lane >= 16) ? 16 : 0;
  v8f c = {};
  for (int kb = 0; kb < FINd; kb += 32) {
    v16bf a, b;
#pragma unroll
    for (int e = 0; e < 8; ++e) {
      a[e]     = (__bf16)arow[kb + koff + e];
      a[e + 8] = (__bf16)arow[kb + koff + 8 + e];
    }
#pragma unroll
    for (int e = 0; e < 16; ++e) b[e] = (__bf16)W[(size_t)(kb + koff + e) * FOUTd + j];
    c = __builtin_amdgcn_wmma_f32_16x16x32_bf16(false, a, false, b, (short)0, c, false, false);
  }
#pragma unroll
  for (int v = 0; v < 8; ++v) {
    int i = mt * 16 + v + ((lane >= 16) ? 8 : 0);
    Wh[(size_t)i * FOUTd + j] = c[v];
  }
}

__global__ void k_whp(const float* __restrict__ Wh, __bf16* __restrict__ whp) {
  int idx = blockIdx.x * blockDim.x + threadIdx.x;
  if (idx >= NTd * FOUTd) return;
  int i = idx / FOUTd;
  whp[idx] = (i < Nn) ? (__bf16)Wh[idx] : (__bf16)0.f;
}

__global__ void k_f1f2(const float* __restrict__ Wh, const float* __restrict__ a,
                       float* __restrict__ f1, float* __restrict__ f2) {
  int i = blockIdx.x * blockDim.x + threadIdx.x;
  if (i >= Nn) return;
  const float* r = Wh + (size_t)i * FOUTd;
  float s1 = 0.f, s2 = 0.f;
  for (int f = 0; f < FOUTd; ++f) { s1 += r[f] * a[f]; s2 += r[f] * a[FOUTd + f]; }
  f1[i] = s1; f2[i] = s2;
}

// ---------------- tiled WMMA GEMM #1: z = sqrt(max(sq_i+sq_j-2*h@h^T, 0)) ----------------
// Both tiles TDM-staged (double-buffered) from the bf16 copy of h.
__global__ void k_gram(const __bf16* __restrict__ hb, const float* __restrict__ sq,
                       float* __restrict__ zP) {
  __shared__ __align__(16) __bf16 As[2][128 * 32];
  __shared__ __align__(16) __bf16 Bs[2][128 * 32];
  int tid = threadIdx.x;
  int bm0 = blockIdx.x * 128, bn0 = blockIdx.y * 128;
  int w = tid >> 5, lane = tid & 31;
  int wm = w >> 1, wn = w & 1;
  int hi8 = (lane >= 16) ? 8 : 0;
  int koffB = (lane >= 16) ? 16 : 0;
  const int NK = FINd / 32;
  v8f acc[2][4] = {};
#if USE_TDM
  if (w == 0) {
    tdm_issue(hb + (size_t)bm0 * FINd, (unsigned)(uintptr_t)&As[0][0], FINd);
    tdm_issue(hb + (size_t)bn0 * FINd, (unsigned)(uintptr_t)&Bs[0][0], FINd);
  }
  for (int it = 0; it < NK; ++it) {
    int cur = it & 1;
    if (it + 1 < NK) {
      if (w == 0) {
        int kb = (it + 1) * 32;
        tdm_issue(hb + (size_t)bm0 * FINd + kb, (unsigned)(uintptr_t)&As[cur ^ 1][0], FINd);
        tdm_issue(hb + (size_t)bn0 * FINd + kb, (unsigned)(uintptr_t)&Bs[cur ^ 1][0], FINd);
        __builtin_amdgcn_s_wait_tensorcnt(2);
      }
    } else {
      if (w == 0) __builtin_amdgcn_s_wait_tensorcnt(0);
    }
    __syncthreads();
    mma_step(&As[cur][0], &Bs[cur][0], wm, wn, lane, hi8, koffB, acc);
    __syncthreads();
  }
#else
  for (int it = 0; it < NK; ++it) {
    int kb = it * 32;
    __syncthreads();
#pragma unroll
    for (int r = 0; r < 4; ++r) {
      int u = tid + 256 * r;
      int row = u >> 3, kpos = (u & 7) * 4;
      *(uint2*)&As[0][row * 32 + kpos] =
          *(const uint2*)&hb[(size_t)(bm0 + row) * FINd + kb + kpos];
      *(uint2*)&Bs[0][row * 32 + kpos] =
          *(const uint2*)&hb[(size_t)(bn0 + row) * FINd + kb + kpos];
    }
    __syncthreads();
    mma_step(&As[0][0], &Bs[0][0], wm, wn, lane, hi8, koffB, acc);
  }
#endif
#pragma unroll
  for (int ms = 0; ms < 2; ++ms) {
#pragma unroll
    for (int ns = 0; ns < 4; ++ns) {
      int j = bn0 + wn * 64 + ns * 16 + (lane & 15);
      float sj = sq[j];
#pragma unroll
      for (int v = 0; v < 8; ++v) {
        int i = bm0 + wm * 32 + ms * 16 + v + hi8;
        float d2 = sq[i] + sj - 2.0f * acc[ms][ns][v];
        zP[(size_t)i * NTd + j] = sqrtf(fmaxf(d2, 0.0f));
      }
    }
  }
}

// per-row top-6 smallest; one wave per row, coalesced lane-strided scan + LDS merge
__global__ void k_topk(const float* __restrict__ zP, int* __restrict__ knn_i,
                       float* __restrict__ knn_d, float* __restrict__ dsec) {
  __shared__ float sv[8 * 32 * TOPKd];
  __shared__ int   si[8 * 32 * TOPKd];
  int tid = threadIdx.x, w = tid >> 5, lane = tid & 31;
  int row = blockIdx.x * 8 + w;
  const float* rp = zP + (size_t)row * NTd;
  float vals[TOPKd]; int idxs[TOPKd];
#pragma unroll
  for (int q = 0; q < TOPKd; ++q) { vals[q] = FLT_MAX; idxs[q] = 0x7FFFFFFF; }
  for (int j = lane; j < Nn; j += 32) {
    float v = rp[j];
    if (v < vals[TOPKd - 1]) {
      int p = TOPKd - 1;
      while (p > 0 && v < vals[p - 1]) { vals[p] = vals[p - 1]; idxs[p] = idxs[p - 1]; --p; }
      vals[p] = v; idxs[p] = j;
    }
  }
  int base = (w * 32 + lane) * TOPKd;
#pragma unroll
  for (int q = 0; q < TOPKd; ++q) { sv[base + q] = vals[q]; si[base + q] = idxs[q]; }
  __syncthreads();
  if (lane == 0) {
    float mv[TOPKd]; int mi[TOPKd];
#pragma unroll
    for (int q = 0; q < TOPKd; ++q) { mv[q] = FLT_MAX; mi[q] = 0x7FFFFFFF; }
    int wb = w * 32 * TOPKd;
    for (int c = 0; c < 32 * TOPKd; ++c) {
      float v = sv[wb + c]; int id = si[wb + c];
      if (v < mv[TOPKd - 1] || (v == mv[TOPKd - 1] && id < mi[TOPKd - 1])) {
        int p = TOPKd - 1;
        while (p > 0 && (v < mv[p - 1] || (v == mv[p - 1] && id < mi[p - 1]))) {
          mv[p] = mv[p - 1]; mi[p] = mi[p - 1]; --p;
        }
        mv[p] = v; mi[p] = id;
      }
    }
    dsec[row] = mv[1];
    for (int q = 0; q < KNNd; ++q) {
      knn_i[row * KNNd + q] = mi[1 + q];
      knn_d[row * KNNd + q] = mv[1 + q];
    }
  }
}

__global__ void k_dc(const float* __restrict__ dsec, float* __restrict__ dc) {
  __shared__ float sh[1024];
  int tid = threadIdx.x;
  float s = 0.f;
  for (int i = tid; i < Nn; i += 1024) s += dsec[i];
  sh[tid] = s; __syncthreads();
  for (int off = 512; off > 0; off >>= 1) {
    if (tid < off) sh[tid] += sh[tid + off];
    __syncthreads();
  }
  if (tid == 0) dc[0] = sh[0] / (float)Nn;
}

// ---------------- tiny k-means (3 centers, 10 iters) ----------------

__global__ void k_copyrow(const float* __restrict__ h, const int* __restrict__ idxp,
                          int use_idx, float* __restrict__ dst) {
  int f = blockIdx.x * blockDim.x + threadIdx.x;
  if (f >= FINd) return;
  int r = use_idx ? idxp[0] : 0;
  dst[f] = h[(size_t)r * FINd + f];
}

__global__ void k_dist2(const float* __restrict__ h, const float* __restrict__ c,
                        float* __restrict__ dmin, int mode) {
  int i = blockIdx.x * blockDim.x + threadIdx.x;
  if (i >= Nn) return;
  const float* r = h + (size_t)i * FINd;
  float s = 0.f;
  for (int f = 0; f < FINd; ++f) { float d = r[f] - c[f]; s += d * d; }
  dmin[i] = mode ? fminf(dmin[i], s) : s;
}

__global__ void k_argmax(const float* __restrict__ d, int* __restrict__ outp) {
  __shared__ float sv[1024]; __shared__ int si[1024];
  int tid = threadIdx.x;
  float bv = -FLT_MAX; int bi = 0;
  for (int i = tid; i < Nn; i += 1024) {
    float v = d[i];
    if (v > bv) { bv = v; bi = i; }
  }
  sv[tid] = bv; si[tid] = bi; __syncthreads();
  for (int off = 512; off > 0; off >>= 1) {
    if (tid < off) {
      if (sv[tid + off] > sv[tid] ||
          (sv[tid + off] == sv[tid] && si[tid + off] < si[tid])) {
        sv[tid] = sv[tid + off]; si[tid] = si[tid + off];
      }
    }
    __syncthreads();
  }
  if (tid == 0) outp[0] = si[0];
}

__global__ void k_zero(float* __restrict__ newc, float* __restrict__ counts) {
  int i = blockIdx.x * blockDim.x + threadIdx.x;
  if (i < CCd * FINd) newc[i] = 0.f;
  if (i < CCd) counts[i] = 0.f;
}

__global__ void k_assign(const float* __restrict__ h, const float* __restrict__ cen,
                         float* __restrict__ newc, float* __restrict__ counts) {
  int i = blockIdx.x * blockDim.x + threadIdx.x;
  if (i >= Nn) return;
  const float* r = h + (size_t)i * FINd;
  float best = FLT_MAX; int bc = 0;
  for (int c = 0; c < CCd; ++c) {
    float s = 0.f;
    for (int f = 0; f < FINd; ++f) { float d = r[f] - cen[c * FINd + f]; s += d * d; }
    if (s < best) { best = s; bc = c; }
  }
  atomicAdd(&counts[bc], 1.f);
  for (int f = 0; f < FINd; ++f) atomicAdd(&newc[bc * FINd + f], r[f]);
}

__global__ void k_update(float* __restrict__ cen, const float* __restrict__ newc,
                         const float* __restrict__ counts) {
  int i = blockIdx.x * blockDim.x + threadIdx.x;
  if (i >= CCd * FINd) return;
  cen[i] = newc[i] / fmaxf(counts[i / FINd], 1.f);
}

__global__ void k_d1t(const float* __restrict__ h, const float* __restrict__ cen,
                      const float* __restrict__ dcp, float* __restrict__ d1t) {
  int i = blockIdx.x * blockDim.x + threadIdx.x;
  if (i >= Nn) return;
  const float* r = h + (size_t)i * FINd;
  float d[CCd];
  for (int c = 0; c < CCd; ++c) {
    float s = 0.f;
    for (int f = 0; f < FINd; ++f) { float df = r[f] - cen[c * FINd + f]; s += df * df; }
    d[c] = sqrtf(fmaxf(s, 0.f));
  }
  float dc = dcp[0], near1 = d[1];
  for (int c = 0; c < CCd; ++c) {
    float v = (d[c] != 0.f) ? dc * near1 / (d[c] * d[c]) : d[c];
    d1t[i * CCd + c] = v - EPSd;
  }
}

// build g (bf16): 8 elements per thread, one b128 store
__global__ void k_gbuild(const float* __restrict__ d1t, __bf16* __restrict__ g) {
  size_t idx = (size_t)blockIdx.x * blockDim.x + threadIdx.x;   // NTd*NTd/8 threads
  int i = (int)(idx / (NTd / 8));
  int jb = (int)(idx % (NTd / 8)) * 8;
  Pack8 pk;
#pragma unroll
  for (int e = 0; e < 8; ++e) {
    int j = jb + e;
    float v;
    if (i < Nn && j < Nn)       v = -EPSd;
    else if (i < Nn && j < NP)  v = d1t[i * CCd + (j - Nn)];
    else if (i < NP && j < Nn)  v = d1t[j * CCd + (i - Nn)];
    else if (i < NP && j < NP)  v = NEGd;
    else                        v = 0.f;
    pk.h[e] = (__bf16)v;
  }
  *(uint4*)&g[(size_t)i * NTd + jb] = pk.u4;
}

__global__ void k_gscatter(const int* __restrict__ knn_i, const float* __restrict__ knn_d,
                           const float* __restrict__ dsec, const float* __restrict__ dcp,
                           __bf16* __restrict__ g) {
  int i = blockIdx.x * blockDim.x + threadIdx.x;
  if (i >= Nn) return;
  int mincol = knn_i[i * KNNd];
  for (int q = 1; q < KNNd; ++q) mincol = min(mincol, knn_i[i * KNNd + q]);
  float dc = dcp[0], ns = dsec[i];
  for (int q = 0; q < KNNd; ++q) {
    int j = knn_i[i * KNNd + q];
    float zv = knn_d[i * KNNd + q];
    long flat = (long)i * Nn + (long)j;
    if (zv != 0.f && flat < (long)mincol)
      g[(size_t)i * NTd + j] = (__bf16)(dc * ns / (zv * zv) - EPSd);
  }
}

// materialize att_pre TRANSPOSED (apT[j][k] = att_pre[k][j]) so the B tile of the
// big GEMM is a plain strided 2D tile -> TDM-stageable with no ALU transpose.
__global__ void k_attpreT(const float* __restrict__ f1, const float* __restrict__ f2,
                          const float* __restrict__ t1, const float* __restrict__ t2,
                          __bf16* __restrict__ apT) {
  size_t idx = (size_t)blockIdx.x * blockDim.x + threadIdx.x;   // NTd*NTd/8 threads
  int j  = (int)(idx / (NTd / 8));          // output row = att_pre column
  int kb = (int)(idx % (NTd / 8)) * 8;      // att_pre row (K) block
  float f2j = (j < Nn) ? f2[j] : 0.f;
  Pack8 pk;
#pragma unroll
  for (int e = 0; e < 8; ++e) {
    int k = kb + e;
    float val;
    if (j < Nn) {
      if (k < Nn)      { float s = f1[k] + f2j; val = (s > 0.f) ? s : ALPHAd * s; }
      else if (k < NP)   val = t2[(k - Nn) * Nn + j];
      else               val = 0.f;
    } else if (j < NP && k < Nn) {
      val = t1[k * CCd + (j - Nn)];
    } else {
      val = 0.f;        // zeros(CC,CC) corner + padding
    }
    pk.h[e] = (__bf16)val;
  }
  *(uint4*)&apT[(size_t)j * NTd + kb] = pk.u4;
}

// ---------------- tiled WMMA GEMM #2 (dominant): P = g @ att_pre ----------------
// Both tiles TDM-staged, double-buffered: DMA of step k+1 overlaps WMMAs of step k.
__global__ void k_pgemm(const __bf16* __restrict__ Ag, const __bf16* __restrict__ BgT,
                        float* __restrict__ P) {
  __shared__ __align__(16) __bf16 As[2][128 * 32];   // [m][k]
  __shared__ __align__(16) __bf16 Bs[2][128 * 32];   // [n][k]
  int tid = threadIdx.x;
  int bm0 = blockIdx.x * 128, bn0 = blockIdx.y * 128;
  int w = tid >> 5, lane = tid & 31;
  int wm = w >> 1, wn = w & 1;
  int hi8 = (lane >= 16) ? 8 : 0;
  int koffB = (lane >= 16) ? 16 : 0;
  const int NK = NTd / 32;
  v8f acc[2][4] = {};
#if USE_TDM
  if (w == 0) {
    tdm_issue(Ag + (size_t)bm0 * NTd, (unsigned)(uintptr_t)&As[0][0], NTd);
    tdm_issue(BgT + (size_t)bn0 * NTd, (unsigned)(uintptr_t)&Bs[0][0], NTd);
  }
  for (int it = 0; it < NK; ++it) {
    int cur = it & 1;
    if (it + 1 < NK) {
      if (w == 0) {
        int kb = (it + 1) * 32;
        tdm_issue(Ag + (size_t)bm0 * NTd + kb, (unsigned)(uintptr_t)&As[cur ^ 1][0], NTd);
        tdm_issue(BgT + (size_t)bn0 * NTd + kb, (unsigned)(uintptr_t)&Bs[cur ^ 1][0], NTd);
        __builtin_amdgcn_s_wait_tensorcnt(2);   // current pair landed; next pair in flight
      }
    } else {
      if (w == 0) __builtin_amdgcn_s_wait_tensorcnt(0);
    }
    __syncthreads();
    mma_step(&As[cur][0], &Bs[cur][0], wm, wn, lane, hi8, koffB, acc);
    __syncthreads();
  }
#else
  for (int it = 0; it < NK; ++it) {
    int kb = it * 32;
    __syncthreads();
#pragma unroll
    for (int r = 0; r < 4; ++r) {
      int u = tid + 256 * r;
      int row = u >> 3, kpos = (u & 7) * 4;
      *(uint2*)&As[0][row * 32 + kpos] =
          *(const uint2*)&Ag[(size_t)(bm0 + row) * NTd + kb + kpos];
      *(uint2*)&Bs[0][row * 32 + kpos] =
          *(const uint2*)&BgT[(size_t)(bn0 + row) * NTd + kb + kpos];
    }
    __syncthreads();
    mma_step(&As[0][0], &Bs[0][0], wm, wn, lane, hi8, koffB, acc);
  }
#endif
#pragma unroll
  for (int ms = 0; ms < 2; ++ms) {
#pragma unroll
    for (int ns = 0; ns < 4; ++ns) {
      int j = bn0 + wn * 64 + ns * 16 + (lane & 15);
#pragma unroll
      for (int v = 0; v < 8; ++v) {
        int i = bm0 + wm * 32 + ms * 16 + v + hi8;
        P[(size_t)i * NTd + j] = acc[ms][ns][v];
      }
    }
  }
}

// row softmax over cols [0,NP), write bf16 att with zero padding
__global__ void k_softmax(const float* __restrict__ P, __bf16* __restrict__ att) {
  int i = blockIdx.x, tid = threadIdx.x;
  __shared__ float sh[256];
  __bf16* arow = att + (size_t)i * NTd;
  if (i >= NP) {
    for (int j = tid; j < NTd; j += 256) arow[j] = (__bf16)0.f;
    return;
  }
  const float* prow = P + (size_t)i * NTd;
  float m = -FLT_MAX;
  for (int j = tid; j < NP; j += 256) m = fmaxf(m, prow[j]);
  sh[tid] = m; __syncthreads();
  for (int off = 128; off > 0; off >>= 1) {
    if (tid < off) sh[tid] = fmaxf(sh[tid], sh[tid + off]);
    __syncthreads();
  }
  m = sh[0]; __syncthreads();
  float s = 0.f;
  for (int j = tid; j < NP; j += 256) s += expf(prow[j] - m);
  sh[tid] = s; __syncthreads();
  for (int off = 128; off > 0; off >>= 1) {
    if (tid < off) sh[tid] += sh[tid + off];
    __syncthreads();
  }
  s = sh[0]; __syncthreads();
  float inv = 1.f / s;
  for (int j = tid; j < NTd; j += 256)
    arow[j] = (j < NP) ? (__bf16)(expf(prow[j] - m) * inv) : (__bf16)0.f;
}

// out = elu(att @ Wh_pad); A fragments via two b128 global loads
__global__ void k_outgemm(const __bf16* __restrict__ att, const __bf16* __restrict__ whp,
                          float* __restrict__ out) {
  const int mtiles = NTd / 16;  // 264
  int t = blockIdx.x * 4 + (threadIdx.x >> 5);
  if (t >= mtiles * 4) return;
  int lane = threadIdx.x & 31;
  int mt = t >> 2, ntile = t & 3;
  const __bf16* arow = att + (size_t)(mt * 16 + (lane & 15)) * NTd;
  int j = ntile * 16 + (lane & 15);
  int hi8 = (lane >= 16) ? 8 : 0;
  int koff = (lane >= 16) ? 16 : 0;
  v8f c = {};
  for (int kb = 0; kb < NTd; kb += 32) {
    FragU fa;
    fa.u[0] = *(const v4u*)&arow[kb + koff];
    fa.u[1] = *(const v4u*)&arow[kb + koff + 8];
    v16bf b;
#pragma unroll
    for (int e = 0; e < 16; ++e) b[e] = whp[(size_t)(kb + koff + e) * FOUTd + j];
    c = __builtin_amdgcn_wmma_f32_16x16x32_bf16(false, fa.v, false, b, (short)0, c, false, false);
  }
#pragma unroll
  for (int v = 0; v < 8; ++v) {
    int i = mt * 16 + v + ((lane >= 16) ? 8 : 0);
    if (i < NP) {
      float x = c[v];
      out[(size_t)i * FOUTd + j] = (x > 0.f) ? x : (expf(x) - 1.f);
    }
  }
}

// ---------------- host launch ----------------

extern "C" void kernel_launch(void* const* d_in, const int* in_sizes, int n_in,
                              void* d_out, int out_size, void* d_ws, size_t ws_size,
                              hipStream_t stream) {
  const float* h  = (const float*)d_in[0];
  // d_in[1] = adj (unused by the reference forward)
  const float* W  = (const float*)d_in[2];
  const float* a  = (const float*)d_in[3];
  const float* t1 = (const float*)d_in[4];
  const float* t2 = (const float*)d_in[5];
  float* out = (float*)d_out;

  char* p = (char*)d_ws;
  auto alloc = [&](size_t bytes) -> char* {
    char* r = p;
    p += (bytes + 255) & ~(size_t)255;
    return r;
  };
  float*  zP    = (float*)alloc((size_t)NTd * NTd * 4);   // z, then reused as P
  __bf16* gbf   = (__bf16*)alloc((size_t)NTd * NTd * 2);
  __bf16* apT   = (__bf16*)alloc((size_t)NTd * NTd * 2);  // att_pre transposed
  __bf16* attbf = (__bf16*)alloc((size_t)NTd * NTd * 2);
  __bf16* hb    = (__bf16*)alloc((size_t)Nn * FINd * 2);
  float*  Wh    = (float*)alloc((size_t)Nn * FOUTd * 4);
  __bf16* whp   = (__bf16*)alloc((size_t)NTd * FOUTd * 2);
  float*  f1    = (float*)alloc((size_t)Nn * 4);
  float*  f2    = (float*)alloc((size_t)Nn * 4);
  float*  sq    = (float*)alloc((size_t)Nn * 4);
  int*    knn_i = (int*)alloc((size_t)Nn * KNNd * 4);
  float*  knn_d = (float*)alloc((size_t)Nn * KNNd * 4);
  float*  dsec  = (float*)alloc((size_t)Nn * 4);
  float*  dcv   = (float*)alloc(256);
  float*  cen   = (float*)alloc((size_t)CCd * FINd * 4);
  float*  newc  = (float*)alloc((size_t)CCd * FINd * 4);
  float*  cnts  = (float*)alloc(256);
  float*  dmin  = (float*)alloc((size_t)Nn * 4);
  int*    amax  = (int*)alloc(256);
  float*  d1t   = (float*)alloc((size_t)Nn * CCd * 4);

  k_sq<<<16, 256, 0, stream>>>(h, sq);
  k_hb<<<(Nn * FINd) / 8 / 256, 256, 0, stream>>>(h, hb);
  k_whgemm<<<256, 128, 0, stream>>>(h, W, Wh);
  k_whp<<<(NTd * FOUTd) / 256, 256, 0, stream>>>(Wh, whp);
  k_f1f2<<<16, 256, 0, stream>>>(Wh, a, f1, f2);
  { dim3 grid(Nn / 128, Nn / 128);
    k_gram<<<grid, 256, 0, stream>>>(hb, sq, zP); }
  k_topk<<<Nn / 8, 256, 0, stream>>>(zP, knn_i, knn_d, dsec);
  k_dc<<<1, 1024, 0, stream>>>(dsec, dcv);

  // k-means init: c0 = x[0]; c1 = farthest from c0; c2 = farthest from {c0,c1}
  k_copyrow<<<2, 256, 0, stream>>>(h, amax, 0, cen);
  k_dist2<<<16, 256, 0, stream>>>(h, cen, dmin, 0);
  k_argmax<<<1, 1024, 0, stream>>>(dmin, amax);
  k_copyrow<<<2, 256, 0, stream>>>(h, amax, 1, cen + FINd);
  k_dist2<<<16, 256, 0, stream>>>(h, cen + FINd, dmin, 1);
  k_argmax<<<1, 1024, 0, stream>>>(dmin, amax);
  k_copyrow<<<2, 256, 0, stream>>>(h, amax, 1, cen + 2 * FINd);
  for (int it = 0; it < 10; ++it) {
    k_zero<<<(CCd * FINd + 255) / 256, 256, 0, stream>>>(newc, cnts);
    k_assign<<<16, 256, 0, stream>>>(h, cen, newc, cnts);
    k_update<<<(CCd * FINd + 255) / 256, 256, 0, stream>>>(cen, newc, cnts);
  }

  k_d1t<<<16, 256, 0, stream>>>(h, cen, dcv, d1t);
  k_gbuild<<<(unsigned)(((size_t)NTd * NTd) / 8 / 256), 256, 0, stream>>>(d1t, gbf);
  k_gscatter<<<16, 256, 0, stream>>>(knn_i, knn_d, dsec, dcv, gbf);
  k_attpreT<<<(unsigned)(((size_t)NTd * NTd) / 8 / 256), 256, 0, stream>>>(f1, f2, t1, t2, apT);
  { dim3 grid(NTd / 128, NTd / 128);
    k_pgemm<<<grid, 256, 0, stream>>>(gbf, apT, zP); }
  k_softmax<<<NTd, 256, 0, stream>>>(zP, attbf);
  k_outgemm<<<(NTd / 16), 128, 0, stream>>>(attbf, whp, out);
}
